// RegularTransformer_45251775430660
// MI455X (gfx1250) — compile-verified, matrix operand order checked by reading, not run on hardware
//
#include <hip/hip_runtime.h>
#include <hip/hip_bf16.h>
#include <math.h>

typedef __bf16 bhalf;
typedef __attribute__((ext_vector_type(16))) __bf16 v16bf;
typedef __attribute__((ext_vector_type(8)))  __bf16 v8bf;
typedef __attribute__((ext_vector_type(8)))  float  v8f;
typedef int v4i_vs __attribute__((vector_size(16)));   // matches builtin proto

#define LAYERS 4
#define DMODEL 512
#define NHEADS 8
#define HDIM   64
#define DFF    2048
#define BATCH  8
#define NTOK   1024
#define ROWS   (BATCH * NTOK)   // 8192

// Async global->LDS path (ASYNCcnt-tracked), if the toolchain declares it.
#if defined(__has_builtin)
#if __has_builtin(__builtin_amdgcn_global_load_async_to_lds_b128)
#define USE_ASYNC_LDS 1
#endif
#if __has_builtin(__builtin_amdgcn_s_wait_asynccnt)
#define HAVE_WAIT_ASYNC 1
#endif
#endif

__device__ inline void async_copy16(const bhalf* g, bhalf* l) {
#if defined(USE_ASYNC_LDS)
  __builtin_amdgcn_global_load_async_to_lds_b128(
      (__attribute__((address_space(1))) v4i_vs*)g,
      (__attribute__((address_space(3))) v4i_vs*)l, 0, 0);
#else
  *(v8bf*)l = *(const v8bf*)g;
#endif
}

__device__ inline void wait_async0() {
#if defined(USE_ASYNC_LDS)
#if defined(HAVE_WAIT_ASYNC)
  __builtin_amdgcn_s_wait_asynccnt(0);
#else
  asm volatile("s_wait_asynccnt 0x0" ::: "memory");
#endif
#endif
}

// ---------------------------------------------------------------------------
// D = A(16x32 bf16) * B(32x16 bf16) + C(16x16 f32), wave32 WMMA
// ---------------------------------------------------------------------------
__device__ inline v8f wmma_bf(v16bf a, v16bf b, v8f c) {
  return __builtin_amdgcn_wmma_f32_16x16x32_bf16(
      /*neg_a=*/false, a, /*neg_b=*/false, b,
      /*c_mod=*/(short)0, c, /*reuse_a=*/false, /*reuse_b=*/false);
}

// Load a 16x32 bf16 fragment (A layout; B symmetric with N as major dim)
// from a buffer laid out [major][k], k contiguous, stride `ld`.
// ISA layout (CDNA5 ISA): lane L holds major = L&15; lanes 0-15 carry
// K={0..7,16..23}, lanes 16-31 carry K={8..15,24..31}; each half is a
// contiguous 8xbf16 = 16-byte load.
__device__ inline v16bf load_frag(const bhalf* base, int ld, int major0, int k0) {
  const int lane = threadIdx.x & 31;
  const int hi   = lane >> 4;
  const bhalf* p = base + (size_t)(major0 + (lane & 15)) * ld + k0 + hi * 8;
  union { v16bf v; v8bf h[2]; } u;
  u.h[0] = *(const v8bf*)p;
  u.h[1] = *(const v8bf*)(p + 16);
  return u.v;
}

// ---------------------------------------------------------------------------
// fp32 -> bf16 conversion (weights)
// ---------------------------------------------------------------------------
__global__ void cvt_kernel(const float* __restrict__ s, bhalf* __restrict__ d, int n) {
  int i = blockIdx.x * blockDim.x + threadIdx.x;
  if (i < n) d[i] = (bhalf)s[i];
}

// (B,C,H,W) -> (B,N,D) tokens
__global__ void to_tokens_kernel(const float* __restrict__ x, float* __restrict__ t) {
  int idx = blockIdx.x * blockDim.x + threadIdx.x;
  if (idx >= ROWS * DMODEL) return;
  int c = idx & (DMODEL - 1);
  int n = (idx >> 9) & (NTOK - 1);
  int b = idx >> 19;
  t[idx] = x[((size_t)b * DMODEL + c) * NTOK + n];
}

// (B,N,D) tokens -> (B,C,H,W)
__global__ void from_tokens_kernel(const float* __restrict__ t, float* __restrict__ out) {
  int idx = blockIdx.x * blockDim.x + threadIdx.x;
  if (idx >= ROWS * DMODEL) return;
  int n = idx & (NTOK - 1);
  int c = (idx >> 10) & (DMODEL - 1);
  int b = idx >> 19;
  out[idx] = t[((size_t)b * NTOK + n) * DMODEL + c];
}

// ---------------------------------------------------------------------------
// LayerNorm (+ optional sinusoidal PE), fp32 in -> bf16 out.
// One wave per row of 512; wave32 shuffle reductions.
// ---------------------------------------------------------------------------
__global__ __launch_bounds__(256) void ln_kernel(
    const float* __restrict__ t, const float* __restrict__ gam,
    const float* __restrict__ bet, bhalf* __restrict__ out, int addPE) {
  const int lane = threadIdx.x & 31;
  const int wid  = threadIdx.x >> 5;
  const int row  = blockIdx.x * 8 + wid;
  const float* x = t + (size_t)row * DMODEL;

  float xv[16];
  float s = 0.f;
#pragma unroll
  for (int j = 0; j < 4; ++j) {
    float4 f4 = *(const float4*)(x + lane * 4 + j * 128);
    xv[j * 4 + 0] = f4.x; xv[j * 4 + 1] = f4.y;
    xv[j * 4 + 2] = f4.z; xv[j * 4 + 3] = f4.w;
    s += f4.x + f4.y + f4.z + f4.w;
  }
#pragma unroll
  for (int m = 1; m < 32; m <<= 1) s += __shfl_xor(s, m, 32);
  const float mu = s * (1.0f / DMODEL);

  float vs = 0.f;
#pragma unroll
  for (int k = 0; k < 16; ++k) { float d = xv[k] - mu; vs += d * d; }
#pragma unroll
  for (int m = 1; m < 32; m <<= 1) vs += __shfl_xor(vs, m, 32);
  const float rsq = rsqrtf(vs * (1.0f / DMODEL) + 1e-5f);

  const int n = row & (NTOK - 1);
#pragma unroll
  for (int j = 0; j < 4; ++j) {
#pragma unroll
    for (int e = 0; e < 4; ++e) {
      int d = lane * 4 + j * 128 + e;
      float y = (xv[j * 4 + e] - mu) * rsq * gam[d] + bet[d];
      if (addPE) {
        // pe = sin/cos( n / 10000^{(d & ~1)/512} ), ln(10000)=9.210340372
        float ang = (float)n * __expf(-(float)(d & ~1) * (9.210340372f / DMODEL));
        y += (d & 1) ? cosf(ang) : sinf(ang);
      }
      out[(size_t)row * DMODEL + d] = (bhalf)y;
    }
  }
}

// ---------------------------------------------------------------------------
// bf16 GEMM: C(MxN) = A(MxK) * B(KxN) [+bias] [GELU] [+fp32 residual io]
// Workgroup tile 128x128, 8 waves (4x2), wave tile 32x64, K-step 32.
// A tile staged via async global->LDS DMA (when available); B tile staged
// transposed (n-major, k-contiguous) so fragments are two ds_load_b128s.
// ---------------------------------------------------------------------------
#define BM 128
#define BN 128
#define BK 32
#define LDT 48  // padded LDS k-stride (96B, keeps 16B chunk alignment)

template <bool BIASF, bool GELUF, bool RESIDF, bool OUTBF>
__global__ __launch_bounds__(256) void gemm_kernel(
    const bhalf* __restrict__ A, const bhalf* __restrict__ B,
    const float* __restrict__ bias, float* __restrict__ io_f,
    bhalf* __restrict__ out_bf, int M, int N, int K) {
  __shared__ alignas(16) bhalf As[BM][LDT];   // [m][k]
  __shared__ alignas(16) bhalf Bs[BN][LDT];   // transposed: [n][k]

  const int tid  = threadIdx.x;
  const int lane = tid & 31;
  const int w    = tid >> 5;
  const int m0   = blockIdx.y * BM;
  const int n0   = blockIdx.x * BN;
  const int mw   = (w >> 1) * 32;
  const int nw   = (w & 1) * 64;

  v8f acc[2][4] = {};

  for (int kk = 0; kk < K; kk += BK) {
    __syncthreads();
    // A tile: 128 rows x 32 k = 512 16B chunks, 2 per thread, async DMA
#pragma unroll
    for (int c = 0; c < 2; ++c) {
      int chunk = tid + c * 256;
      int arow = chunk >> 2, ac = (chunk & 3) * 8;
      async_copy16(A + (size_t)(m0 + arow) * K + kk + ac, &As[arow][ac]);
    }
    // B tile: 32 rows x 128 n, transpose-scatter into Bs[n][k]
#pragma unroll
    for (int c = 0; c < 2; ++c) {
      int chunk = tid + c * 256;
      int brow = chunk >> 4, bc = (chunk & 15) * 8;
      v8bf bv = *(const v8bf*)(B + (size_t)(kk + brow) * N + n0 + bc);
#pragma unroll
      for (int j = 0; j < 8; ++j) Bs[bc + j][brow] = bv[j];
    }
    if (kk + BK < K)
      __builtin_prefetch(B + (size_t)(kk + BK) * N + n0 + (tid & 15) * 8, 0, 1);
    wait_async0();
    __syncthreads();

    v16bf af0 = load_frag(&As[0][0], LDT, mw, 0);
    v16bf af1 = load_frag(&As[0][0], LDT, mw + 16, 0);
#pragma unroll
    for (int t = 0; t < 4; ++t) {
      v16bf bfrag = load_frag(&Bs[0][0], LDT, nw + t * 16, 0);
      acc[0][t] = wmma_bf(af0, bfrag, acc[0][t]);
      acc[1][t] = wmma_bf(af1, bfrag, acc[1][t]);
    }
  }

  // Epilogue. C layout: VGPR r <-> row r + 8*(lane>>4), lane&15 <-> col.
  const int hi = lane >> 4;
#pragma unroll
  for (int t = 0; t < 4; ++t) {
    int cg = n0 + nw + t * 16 + (lane & 15);
    float bv = BIASF ? bias[cg] : 0.0f;
#pragma unroll
    for (int ms = 0; ms < 2; ++ms) {
#pragma unroll
      for (int r = 0; r < 8; ++r) {
        int rg = m0 + mw + ms * 16 + r + 8 * hi;
        float v = acc[ms][t][r] + bv;
        if (GELUF) v = 0.5f * v * (1.0f + erff(v * 0.70710678118f));
        size_t idx = (size_t)rg * N + cg;
        if (OUTBF)       out_bf[idx] = (bhalf)v;
        else if (RESIDF) io_f[idx] = io_f[idx] + v;
        else             io_f[idx] = v;
      }
    }
  }
}

// ---------------------------------------------------------------------------
// Flash attention. Grid (N/64, NHEADS, BATCH), 128 threads = 4 waves.
// Wave owns 16 q rows; K/V tiles of 32 tokens staged in LDS (V transposed).
// qkv layout: [row = b*N+n][3*D] with q|k|v blocks of D, head h at h*64.
// ---------------------------------------------------------------------------
__global__ __launch_bounds__(128) void attn_kernel(
    const bhalf* __restrict__ qkv, bhalf* __restrict__ attno) {
  __shared__ alignas(16) bhalf Ks[32][72];     // [tok][hd]
  __shared__ alignas(16) bhalf Vt[64][48];     // [hd][tok]
  __shared__ alignas(16) bhalf Ps[4][16][48];  // per-wave P scratch [qrow][tok]

  const int tid  = threadIdx.x;
  const int lane = tid & 31;
  const int w    = tid >> 5;
  const int hi   = lane >> 4;
  const int b = blockIdx.z, h = blockIdx.y, qt = blockIdx.x;
  const int qrow0 = qt * 64 + w * 16;

  // Q fragments: A-matrix layout, straight from global (k = hd, 2 chunks)
  v16bf aq[2];
  {
    const bhalf* qb = qkv +
        (size_t)(b * NTOK + qrow0 + (lane & 15)) * (3 * DMODEL) + h * HDIM;
#pragma unroll
    for (int c = 0; c < 2; ++c) {
      union { v16bf v; v8bf hh[2]; } u;
      u.hh[0] = *(const v8bf*)(qb + c * 32 + hi * 8);
      u.hh[1] = *(const v8bf*)(qb + c * 32 + hi * 8 + 16);
      aq[c] = u.v;
    }
  }

  v8f o[4] = {};
  float mrun[8], lrun[8];
#pragma unroll
  for (int r = 0; r < 8; ++r) { mrun[r] = -3.0e38f; lrun[r] = 0.f; }

  for (int kb = 0; kb < NTOK / 32; ++kb) {
    __syncthreads();
    {
      // cooperative K/V stage: 32 toks x 64 hd, 8-elem chunks, 2 per thread
#pragma unroll
      for (int c = 0; c < 2; ++c) {
        int chunk = tid + c * 128;
        int tok = chunk >> 3, hc = (chunk & 7) * 8;
        size_t base = (size_t)(b * NTOK + kb * 32 + tok) * (3 * DMODEL) + h * HDIM + hc;
        *(v8bf*)&Ks[tok][hc] = *(const v8bf*)(qkv + base + DMODEL);
        v8bf vv = *(const v8bf*)(qkv + base + 2 * DMODEL);
#pragma unroll
        for (int j = 0; j < 8; ++j) Vt[hc + j][tok] = vv[j];
      }
    }
    __syncthreads();

    // scores: S(16x32) = Q(16x64) * K^T, two 16-col tiles, K-chunks over hd
    v8f sc[2];
#pragma unroll
    for (int t = 0; t < 2; ++t) {
      v8f s = {};
#pragma unroll
      for (int c = 0; c < 2; ++c) {
        v16bf bk = load_frag(&Ks[0][0], 72, t * 16, c * 32);
        s = wmma_bf(aq[c], bk, s);
      }
      sc[t] = s;
    }

    // online softmax (C layout: row r+8*hi, col lane&15; 16-lane reductions)
    float e0[8], e1[8];
#pragma unroll
    for (int r = 0; r < 8; ++r) {
      float s0 = sc[0][r] * 0.125f, s1 = sc[1][r] * 0.125f;
      float mx = fmaxf(s0, s1);
#pragma unroll
      for (int m = 1; m < 16; m <<= 1) mx = fmaxf(mx, __shfl_xor(mx, m, 32));
      float mN = fmaxf(mrun[r], mx);
      float corr = __expf(mrun[r] - mN);
      mrun[r] = mN;
      float p0 = __expf(s0 - mN), p1 = __expf(s1 - mN);
      float ps = p0 + p1;
#pragma unroll
      for (int m = 1; m < 16; m <<= 1) ps += __shfl_xor(ps, m, 32);
      lrun[r] = lrun[r] * corr + ps;
#pragma unroll
      for (int t = 0; t < 4; ++t) o[t][r] *= corr;
      e0[r] = p0; e1[r] = p1;
    }

    // C-layout -> A-layout refragmentation of P via per-wave LDS
#pragma unroll
    for (int r = 0; r < 8; ++r) {
      Ps[w][r + 8 * hi][lane & 15]        = (bhalf)e0[r];
      Ps[w][r + 8 * hi][16 + (lane & 15)] = (bhalf)e1[r];
    }
    v16bf pa = load_frag(&Ps[w][0][0], 48, 0, 0);
#pragma unroll
    for (int t = 0; t < 4; ++t) {
      v16bf bv = load_frag(&Vt[0][0], 48, t * 16, 0);   // N = hd, K = tok
      o[t] = wmma_bf(pa, bv, o[t]);
    }
  }

  // normalize and store (B,N,h*64+d) in bf16
#pragma unroll
  for (int t = 0; t < 4; ++t) {
#pragma unroll
    for (int r = 0; r < 8; ++r) {
      float v = o[t][r] / lrun[r];
      size_t row = (size_t)b * NTOK + qrow0 + r + 8 * hi;
      attno[row * DMODEL + h * HDIM + t * 16 + (lane & 15)] = (bhalf)v;
    }
  }
}

// ---------------------------------------------------------------------------
// Host orchestration
// ---------------------------------------------------------------------------
extern "C" void kernel_launch(void* const* d_in, const int* in_sizes, int n_in,
                              void* d_out, int out_size, void* d_ws, size_t ws_size,
                              hipStream_t stream) {
  (void)in_sizes; (void)n_in; (void)out_size; (void)ws_size;
  const float* x     = (const float*)d_in[0];
  const float* qkv_w = (const float*)d_in[1];
  const float* lin_w = (const float*)d_in[2];
  const float* lin_b = (const float*)d_in[3];
  const float* fc1_w = (const float*)d_in[4];
  const float* fc1_b = (const float*)d_in[5];
  const float* fc2_w = (const float*)d_in[6];
  const float* fc2_b = (const float*)d_in[7];
  const float* n1_g  = (const float*)d_in[8];
  const float* n1_b  = (const float*)d_in[9];
  const float* n2_g  = (const float*)d_in[10];
  const float* n2_b  = (const float*)d_in[11];
  float* out = (float*)d_out;

  char* ws = (char*)d_ws;
  size_t off = 0;
  auto alloc = [&](size_t bytes) -> void* {
    void* p = ws + off;
    off = (off + bytes + 255) & ~(size_t)255;
    return p;
  };
  bhalf* wqkv  = (bhalf*)alloc((size_t)LAYERS * DMODEL * 3 * DMODEL * 2);
  bhalf* wlin  = (bhalf*)alloc((size_t)LAYERS * DMODEL * DMODEL * 2);
  bhalf* wfc1  = (bhalf*)alloc((size_t)LAYERS * DMODEL * DFF * 2);
  bhalf* wfc2  = (bhalf*)alloc((size_t)LAYERS * DFF * DMODEL * 2);
  float* tbuf  = (float*)alloc((size_t)ROWS * DMODEL * 4);
  bhalf* hs    = (bhalf*)alloc((size_t)ROWS * DMODEL * 2);
  bhalf* qkvb  = (bhalf*)alloc((size_t)ROWS * 3 * DMODEL * 2);
  bhalf* attno = (bhalf*)alloc((size_t)ROWS * DMODEL * 2);
  bhalf* mlp1  = (bhalf*)alloc((size_t)ROWS * DFF * 2);

  int n;
  n = LAYERS * DMODEL * 3 * DMODEL;
  cvt_kernel<<<(n + 255) / 256, 256, 0, stream>>>(qkv_w, wqkv, n);
  n = LAYERS * DMODEL * DMODEL;
  cvt_kernel<<<(n + 255) / 256, 256, 0, stream>>>(lin_w, wlin, n);
  n = LAYERS * DMODEL * DFF;
  cvt_kernel<<<(n + 255) / 256, 256, 0, stream>>>(fc1_w, wfc1, n);
  n = LAYERS * DFF * DMODEL;
  cvt_kernel<<<(n + 255) / 256, 256, 0, stream>>>(fc2_w, wfc2, n);

  to_tokens_kernel<<<(ROWS * DMODEL + 255) / 256, 256, 0, stream>>>(x, tbuf);

  for (int i = 0; i < LAYERS; ++i) {
    // LN1 + PE -> hs (bf16)
    ln_kernel<<<ROWS / 8, 256, 0, stream>>>(
        tbuf, n1_g + i * DMODEL, n1_b + i * DMODEL, hs, 1);
    // qkv = hs @ Wqkv -> bf16
    gemm_kernel<false, false, false, true>
        <<<dim3(3 * DMODEL / BN, ROWS / BM), 256, 0, stream>>>(
            hs, wqkv + (size_t)i * DMODEL * 3 * DMODEL, nullptr, nullptr, qkvb,
            ROWS, 3 * DMODEL, DMODEL);
    // flash attention -> attno (bf16)
    attn_kernel<<<dim3(NTOK / 64, NHEADS, BATCH), 128, 0, stream>>>(qkvb, attno);
    // t += attno @ Wlin + b
    gemm_kernel<true, false, true, false>
        <<<dim3(DMODEL / BN, ROWS / BM), 256, 0, stream>>>(
            attno, wlin + (size_t)i * DMODEL * DMODEL, lin_b + i * DMODEL,
            tbuf, nullptr, ROWS, DMODEL, DMODEL);
    // LN2 -> hs (bf16)
    ln_kernel<<<ROWS / 8, 256, 0, stream>>>(
        tbuf, n2_g + i * DMODEL, n2_b + i * DMODEL, hs, 0);
    // mlp1 = gelu(hs @ Wfc1 + b) -> bf16
    gemm_kernel<true, true, false, true>
        <<<dim3(DFF / BN, ROWS / BM), 256, 0, stream>>>(
            hs, wfc1 + (size_t)i * DMODEL * DFF, fc1_b + i * DFF,
            nullptr, mlp1, ROWS, DFF, DMODEL);
    // t += gelu(mlp1 @ Wfc2 + b)
    gemm_kernel<true, true, true, false>
        <<<dim3(DMODEL / BN, ROWS / BM), 256, 0, stream>>>(
            mlp1, wfc2 + (size_t)i * DFF * DMODEL, fc2_b + i * DMODEL,
            tbuf, nullptr, ROWS, DMODEL, DFF);
  }

  from_tokens_kernel<<<(ROWS * DMODEL + 255) / 256, 256, 0, stream>>>(tbuf, out);
}
